// LCGN_13125420057192
// MI455X (gfx1250) — matrix operand order repaired and checked by reading, not compiled, not run on hardware
//
#include <hip/hip_runtime.h>
#include <cstdint>
#include <cstddef>

typedef __attribute__((ext_vector_type(16))) __bf16 v16bf;
typedef __attribute__((ext_vector_type(8)))  float  v8f;

#define DEV_INLINE __device__ __forceinline__

DEV_INLINE __bf16 f2bf(float f) {
  uint32_t u = __builtin_bit_cast(uint32_t, f);
  uint32_t r = u + 0x7FFFu + ((u >> 16) & 1u);   // round-to-nearest-even
  uint16_t h = (uint16_t)(r >> 16);
  return __builtin_bit_cast(__bf16, h);
}

// LDS byte offset (within workgroup allocation) of a __shared__ pointer:
// generic -> addrspace(3) cast yields the 32-bit LDS offset the async
// instructions' VDST operand expects.
DEV_INLINE uint32_t lds_off(const void* p) {
  return (uint32_t)(uintptr_t)(__attribute__((address_space(3))) const void*)p;
}

// async 16B global -> LDS copy (ASYNCcnt-tracked, per-lane)
DEV_INLINE void async_cp16(uint32_t lds, const void* gptr) {
  asm volatile("global_load_async_to_lds_b128 %0, %1, off"
               :: "v"(lds), "v"((uint64_t)(uintptr_t)gptr)
               : "memory");
}
DEV_INLINE void async_wait0() {
  asm volatile("s_wait_asynccnt 0x0" ::: "memory");
}

// ---------------------------------------------------------------------------
// Generic batched bf16 WMMA GEMM:  C[z] = alpha * A[z]*B[z] + bias, then
// optional per-row-group scale (cmd gating) and elementwise f32 multiplier.
// A: [M,K] row-major (lda).  B: [K,N] row-major (ldb), or [N,K] if BTRANS.
// Tile 64x256x32; 256 threads = 8 wave32 waves; each wave: 32 rows x 64 cols
// = 2 A-frags x 4 B-frags = 8 WMMAs per K-step.  LDS double-buffered; A tile
// (and B tile when BTRANS) staged with GLOBAL_LOAD_ASYNC_TO_LDS_B128 so the
// next tile's fill overlaps the current tile's WMMAs.
// Requires M % 64 == 0, N % 256 == 0, K % 32 == 0 (true for all model shapes).
// ---------------------------------------------------------------------------
template<bool BTRANS>
__global__ __launch_bounds__(256)
void gemm_bf16_wmma(const __bf16* __restrict__ A, int lda, long batchA,
                    const __bf16* __restrict__ B, int ldb, long batchB,
                    const float* __restrict__ bias,
                    const float* __restrict__ rowscale, int rsgroup,
                    const float* __restrict__ mulf, int mul_ld,
                    float* __restrict__ Cf, int ldcf,
                    __bf16* __restrict__ Cb, int ldcb, long batchC,
                    int M, int N, int K, float alpha)
{
  __shared__ __bf16 sA[2][64][40];    // +8 pad: keeps 16B align, spreads banks
  __shared__ __bf16 sB[2][256][40];   // stored transposed: sB[buf][n][k]

  const int tid  = threadIdx.x;
  const int lane = tid & 31;
  const int wave = tid >> 5;          // 0..7
  const int l15  = lane & 15;
  const int hh   = lane >> 4;         // 0..1
  const int mrow0 = (wave & 1) * 32;  // wave's row strip within 64
  const int ncol0 = (wave >> 1) * 64; // wave's col strip within 256

  const long z = blockIdx.z;
  const __bf16* Ab = A + z * batchA;
  const __bf16* Bb = B + z * batchB;

  const int m0 = blockIdx.y * 64;
  const int n0 = blockIdx.x * 256;

  v8f acc[2][4];
  #pragma unroll
  for (int i = 0; i < 2; ++i)
    #pragma unroll
    for (int j = 0; j < 4; ++j)
      #pragma unroll
      for (int r = 0; r < 8; ++r) acc[i][j][r] = 0.0f;

  // ---- stage one 64x32 A tile + 32x256 B tile into buffer `buf` ----
  auto stage = [&](int buf, int kb) {
    { // A: 2048 elems / 256 thr = one async 16B chunk each
      const int r = tid >> 2;
      const int c = (tid & 3) << 3;
      async_cp16(lds_off(&sA[buf][r][c]),
                 Ab + (long)(m0 + r) * lda + (kb + c));
    }
    if (BTRANS) { // B is [N,K] row-major: straight async copy into sB[n][k]
      const __bf16* src = Bb + (long)(n0 + tid) * ldb + kb;
      #pragma unroll
      for (int c4 = 0; c4 < 4; ++c4)
        async_cp16(lds_off(&sB[buf][tid][c4 * 8]), src + c4 * 8);
    } else {      // B is [K,N] row-major: coalesced 64B load, transpose in VGPRs
      const int k  = tid >> 3;            // 0..31
      const int ng = (tid & 7) << 5;      // n group of 32
      const __bf16* src = Bb + (long)(kb + k) * ldb + (n0 + ng);
      union { int4 q[4]; __bf16 e[32]; } u;
      u.q[0] = *reinterpret_cast<const int4*>(src);
      u.q[1] = *reinterpret_cast<const int4*>(src + 8);
      u.q[2] = *reinterpret_cast<const int4*>(src + 16);
      u.q[3] = *reinterpret_cast<const int4*>(src + 24);
      #pragma unroll
      for (int i = 0; i < 32; ++i) sB[buf][ng + i][k] = u.e[i];
    }
  };

  const int nk = K >> 5;
  stage(0, 0);
  async_wait0();
  __syncthreads();

  for (int t = 0; t < nk; ++t) {
    const int cur = t & 1;
    if (t + 1 < nk) stage(cur ^ 1, (t + 1) << 5);   // overlap with compute

    // ---- A fragments (16x32 bf16 layout): lane l<16 row M=l holds K 0..7 /
    //      16..23; lane l>=16 same row holds K 8..15 / 24..31.
    v16bf afr[2];
    #pragma unroll
    for (int i = 0; i < 2; ++i) {
      union { v16bf v; int4 q[2]; } ua;
      const __bf16* ap = &sA[cur][mrow0 + 16 * i + l15][hh * 8];
      ua.q[0] = *reinterpret_cast<const int4*>(ap);
      ua.q[1] = *reinterpret_cast<const int4*>(ap + 16);
      afr[i] = ua.v;
    }
    // ---- 4 B fragments x 2 A fragments = 8 WMMAs
    #pragma unroll
    for (int j = 0; j < 4; ++j) {
      union { v16bf v; int4 q[2]; } ub;
      const __bf16* bp = &sB[cur][ncol0 + 16 * j + l15][hh * 8];
      ub.q[0] = *reinterpret_cast<const int4*>(bp);
      ub.q[1] = *reinterpret_cast<const int4*>(bp + 16);
      #pragma unroll
      for (int i = 0; i < 2; ++i)
        acc[i][j] = __builtin_amdgcn_wmma_f32_16x16x32_bf16(
            false, afr[i], false, ub.v, (short)0, acc[i][j], false, false);
    }

    async_wait0();      // next tile's async fills complete
    __syncthreads();    // all waves done reading `cur`, fills visible
  }

  // ---- epilogue: C/D layout: VGPR r -> row (hh*8 + r), col = lane&15
  const long zc = z * batchC;
  #pragma unroll
  for (int i = 0; i < 2; ++i) {
    const int row_base = m0 + mrow0 + 16 * i + hh * 8;
    #pragma unroll
    for (int j = 0; j < 4; ++j) {
      const int   col = n0 + ncol0 + 16 * j + l15;
      const float bv  = bias ? bias[col] : 0.0f;
      const float sv  = rowscale ? rowscale[(long)(row_base / rsgroup) * N + col] : 1.0f;
      #pragma unroll
      for (int r = 0; r < 8; ++r) {
        const long grow = row_base + r;
        float v = acc[i][j][r] * alpha + bv;
        v *= sv;
        if (mulf) v *= mulf[grow * (long)mul_ld + col];
        if (Cf) Cf[zc + grow * (long)ldcf + col] = v;
        if (Cb) Cb[zc + grow * (long)ldcb + col] = f2bf(v);
      }
    }
  }
}

// ---------------------------------------------------------------------------
// Elementwise / small kernels
// ---------------------------------------------------------------------------
__global__ __launch_bounds__(256)
void cvt_bf16_kernel(const float* __restrict__ in, __bf16* __restrict__ out, long n) {
  long stride = (long)gridDim.x * blockDim.x;
  for (long i = (long)blockIdx.x * blockDim.x + threadIdx.x; i < n; i += stride)
    out[i] = f2bf(in[i]);
}

__global__ __launch_bounds__(256)
void bcast_kernel(const float* __restrict__ src512, float* __restrict__ dst, long n) {
  long stride = (long)gridDim.x * blockDim.x;
  for (long i = (long)blockIdx.x * blockDim.x + threadIdx.x; i < n; i += stride)
    dst[i] = src512[i & 511];
}

// normalize each 1024-float image row and emit bf16
__global__ __launch_bounds__(256)
void norm_kernel(const float* __restrict__ img, __bf16* __restrict__ out) {
  const long row = blockIdx.x;
  const float* r = img + row * 1024;
  __shared__ float red[256];
  float s = 0.0f;
  for (int i = threadIdx.x; i < 1024; i += 256) { float v = r[i]; s += v * v; }
  red[threadIdx.x] = s;
  __syncthreads();
  for (int off = 128; off > 0; off >>= 1) {
    if (threadIdx.x < off) red[threadIdx.x] += red[threadIdx.x + off];
    __syncthreads();
  }
  const float inv = 1.0f / fmaxf(sqrtf(red[0]), 1e-12f);
  for (int i = threadIdx.x; i < 1024; i += 256) out[row * 1024 + i] = f2bf(r[i] * inv);
}

// q_base = elu(q_enc @ W + b), one block per batch row
__global__ __launch_bounds__(256)
void qbase_kernel(const float* __restrict__ qenc, const float* __restrict__ W,
                  const float* __restrict__ b, float* __restrict__ qb) {
  const int bi = blockIdx.x;
  __shared__ float qe[512];
  for (int i = threadIdx.x; i < 512; i += 256) qe[i] = qenc[bi * 512 + i];
  __syncthreads();
  for (int d = threadIdx.x; d < 512; d += 256) {
    float s = b[d];
    for (int k = 0; k < 512; ++k) s += qe[k] * W[k * 512 + d];
    qb[bi * 512 + d] = (s > 0.0f) ? s : (__expf(s) - 1.0f);
  }
}

// per-step: q_cmd, masked softmax attention over L=32, cmd, then cmd->key/val gates
__global__ __launch_bounds__(256)
void cmd_kernel(const float* __restrict__ qb,
                const float* __restrict__ W2, const float* __restrict__ b2,
                const float* __restrict__ lstm, const int* __restrict__ qlen,
                const float* __restrict__ wc2l, const float* __restrict__ bc2l,
                const float* __restrict__ Wpk, const float* __restrict__ bpk,
                const float* __restrict__ Wpv, const float* __restrict__ bpv,
                float* __restrict__ sclK, float* __restrict__ sclV) {
  const int bi = blockIdx.x;
  const int tid = threadIdx.x;
  __shared__ float qbe[512], qc[512], att[32];
  for (int i = tid; i < 512; i += 256) qbe[i] = qb[bi * 512 + i];
  __syncthreads();
  for (int d = tid; d < 512; d += 256) {        // q_cmd
    float s = b2[d];
    for (int k = 0; k < 512; ++k) s += qbe[k] * W2[k * 512 + d];
    qc[d] = s;
  }
  __syncthreads();
  if (tid < 32) {                                // raw_att + mask
    const float* lr = lstm + ((long)bi * 32 + tid) * 512;
    float s = 0.0f;
    for (int d = 0; d < 512; ++d) s += qc[d] * lr[d] * wc2l[d];
    s += bc2l[0];
    if (tid >= qlen[bi]) s = -1e30f;
    att[tid] = s;
  }
  __syncthreads();
  if (tid == 0) {                                // softmax over 32
    float m = att[0];
    for (int l = 1; l < 32; ++l) m = fmaxf(m, att[l]);
    float sum = 0.0f;
    for (int l = 0; l < 32; ++l) { att[l] = __expf(att[l] - m); sum += att[l]; }
    const float inv = 1.0f / sum;
    for (int l = 0; l < 32; ++l) att[l] *= inv;
  }
  __syncthreads();
  for (int d = tid; d < 512; d += 256) {         // cmd (reuse qc)
    float s = 0.0f;
    for (int l = 0; l < 32; ++l) s += att[l] * lstm[((long)bi * 32 + l) * 512 + d];
    qc[d] = s;
  }
  __syncthreads();
  for (int d = tid; d < 512; d += 256) {         // key / val gates
    float sk = bpk[d], sv = bpv[d];
    for (int k = 0; k < 512; ++k) {
      const float c = qc[k];
      sk += c * Wpk[k * 512 + d];
      sv += c * Wpv[k * 512 + d];
    }
    sclK[bi * 512 + d] = sk;
    sclV[bi * 512 + d] = sv;
  }
}

// masked softmax over the 256-wide edge rows, f32 -> bf16 probabilities
__global__ __launch_bounds__(256)
void softmax_kernel(const float* __restrict__ S, __bf16* __restrict__ P,
                    const int* __restrict__ ent) {
  const long row = blockIdx.x;          // b*256 + q
  const int  b = (int)(row >> 8);
  const int  q = (int)(row & 255);
  const int  tid = threadIdx.x;
  __shared__ float red[256];
  const int e = ent[b];
  float v = S[row * 256 + tid];
  if (q >= e || tid >= e) v = -1e30f;
  red[tid] = v;
  __syncthreads();
  for (int off = 128; off > 0; off >>= 1) {
    if (tid < off) red[tid] = fmaxf(red[tid], red[tid + off]);
    __syncthreads();
  }
  const float m = red[0];
  __syncthreads();
  const float x = __expf(v - m);
  red[tid] = x;
  __syncthreads();
  for (int off = 128; off > 0; off >>= 1) {
    if (tid < off) red[tid] += red[tid + off];
    __syncthreads();
  }
  P[row * 256 + tid] = f2bf(x / red[0]);
}

// bf16(x_ctx) into both x_joint[:,512:1024] and mem_in[:,0:512]
__global__ __launch_bounds__(256)
void pack_xctx_kernel(const float* __restrict__ xctx, __bf16* __restrict__ xjoint,
                      __bf16* __restrict__ memin, long n) {
  long stride = (long)gridDim.x * blockDim.x;
  for (long i = (long)blockIdx.x * blockDim.x + threadIdx.x; i < n; i += stride) {
    const long r = i >> 9;
    const int  c = (int)(i & 511);
    const __bf16 v = f2bf(xctx[i]);
    xjoint[r * 1536 + 512 + c] = v;
    memin[r * 1024 + c] = v;
  }
}

// final concat(x_loc, x_ctx) in bf16 (reuses mem_in buffer)
__global__ __launch_bounds__(256)
void pack_final_kernel(const float* __restrict__ xloc, const float* __restrict__ xctx,
                       __bf16* __restrict__ fin, long n) {
  long stride = (long)gridDim.x * blockDim.x;
  for (long i = (long)blockIdx.x * blockDim.x + threadIdx.x; i < n; i += stride) {
    const long r = i >> 9;
    const int  c = (int)(i & 511);
    fin[r * 1024 + c]       = f2bf(xloc[i]);
    fin[r * 1024 + 512 + c] = f2bf(xctx[i]);
  }
}

// ---------------------------------------------------------------------------
// Host orchestration
// ---------------------------------------------------------------------------
static inline void gemm_launch(hipStream_t st,
    const __bf16* A, int lda, long bA,
    const __bf16* B, int ldb, long bB, int btrans,
    const float* bias, const float* rowscale, int rsg,
    const float* mulf, int mulld,
    float* Cf, int ldcf, __bf16* Cb, int ldcb, long bC,
    int M, int N, int K, int batch, float alpha) {
  dim3 g(N / 256, M / 64, batch), blk(256);
  if (btrans)
    gemm_bf16_wmma<true><<<g, blk, 0, st>>>(A, lda, bA, B, ldb, bB, bias,
        rowscale, rsg, mulf, mulld, Cf, ldcf, Cb, ldcb, bC, M, N, K, alpha);
  else
    gemm_bf16_wmma<false><<<g, blk, 0, st>>>(A, lda, bA, B, ldb, bB, bias,
        rowscale, rsg, mulf, mulld, Cf, ldcf, Cb, ldcb, bC, M, N, K, alpha);
}

extern "C" void kernel_launch(void* const* d_in, const int* in_sizes, int n_in,
                              void* d_out, int out_size, void* d_ws, size_t ws_size,
                              hipStream_t stream) {
  (void)in_sizes; (void)n_in; (void)out_size; (void)ws_size;

  const float* images   = (const float*)d_in[0];
  const float* q_enc    = (const float*)d_in[1];
  const float* lstm     = (const float*)d_in[2];
  const int*   qlen     = (const int*)d_in[3];
  const int*   ent      = (const int*)d_in[4];
  const float* initKB_W = (const float*)d_in[5];
  const float* initKB_b = (const float*)d_in[6];
  const float* initMem  = (const float*)d_in[7];
  const float* qIn_W    = (const float*)d_in[8];
  const float* qIn_b    = (const float*)d_in[9];
  const float* qIn2_W   = (const float*)d_in[10];
  const float* qIn2_b   = (const float*)d_in[11];
  const float* c2l_W    = (const float*)d_in[12];
  const float* c2l_b    = (const float*)d_in[13];
  const float* pxl_W    = (const float*)d_in[14];
  const float* pxl_b    = (const float*)d_in[15];
  const float* pxc_W    = (const float*)d_in[16];
  const float* pxc_b    = (const float*)d_in[17];
  const float* qry_W    = (const float*)d_in[18];
  const float* qry_b    = (const float*)d_in[19];
  const float* key_W    = (const float*)d_in[20];
  const float* key_b    = (const float*)d_in[21];
  const float* val_W    = (const float*)d_in[22];
  const float* val_b    = (const float*)d_in[23];
  const float* pk_W     = (const float*)d_in[24];
  const float* pk_b     = (const float*)d_in[25];
  const float* pv_W     = (const float*)d_in[26];
  const float* pv_b     = (const float*)d_in[27];
  const float* mem_W    = (const float*)d_in[28];
  const float* mem_b    = (const float*)d_in[29];
  const float* comb_W   = (const float*)d_in[30];
  const float* comb_b   = (const float*)d_in[31];

  const long M = 32768;                 // B*N rows
  const float SCALE = 0.044194173824159216f;  // 1/sqrt(512)

  // ---- carve workspace ----
  size_t off = 0;
  char* base = (char*)d_ws;
  auto carve = [&](size_t bytes) -> char* {
    char* p = base + off;
    off += (bytes + 255) & ~(size_t)255;
    return p;
  };
  __bf16* wInit = (__bf16*)carve((size_t)1024 * 512 * 2);
  __bf16* wPxl  = (__bf16*)carve((size_t)512 * 512 * 2);
  __bf16* wPxc  = (__bf16*)carve((size_t)512 * 512 * 2);
  __bf16* wQ    = (__bf16*)carve((size_t)1536 * 512 * 2);
  __bf16* wK    = (__bf16*)carve((size_t)1536 * 512 * 2);
  __bf16* wV    = (__bf16*)carve((size_t)1536 * 512 * 2);
  __bf16* wMem  = (__bf16*)carve((size_t)1024 * 512 * 2);
  __bf16* wComb = (__bf16*)carve((size_t)1024 * 512 * 2);
  __bf16* Aimg  = (__bf16*)carve((size_t)M * 1024 * 2);
  float*  xloc  = (float*) carve((size_t)M * 512 * 4);
  float*  xctx  = (float*) carve((size_t)M * 512 * 4);
  __bf16* xjnt  = (__bf16*)carve((size_t)M * 1536 * 2);
  float*  pxl   = (float*) carve((size_t)M * 512 * 4);
  __bf16* Qb    = (__bf16*)carve((size_t)M * 512 * 2);
  __bf16* Kb    = (__bf16*)carve((size_t)M * 512 * 2);
  __bf16* Vb    = (__bf16*)carve((size_t)M * 512 * 2);
  float*  Sbuf  = (float*) carve((size_t)128 * 256 * 256 * 4);
  __bf16* Pbuf  = (__bf16*)carve((size_t)128 * 256 * 256 * 2);
  __bf16* memin = (__bf16*)carve((size_t)M * 1024 * 2);
  float*  qbase = (float*) carve((size_t)128 * 512 * 4);
  float*  sclK  = (float*) carve((size_t)128 * 512 * 4);
  float*  sclV  = (float*) carve((size_t)128 * 512 * 4);

  const int EW = 8192;  // grid-stride elementwise blocks

  // ---- weight conversion (once per call) ----
  cvt_bf16_kernel<<<512, 256, 0, stream>>>(initKB_W, wInit, (long)1024 * 512);
  cvt_bf16_kernel<<<512, 256, 0, stream>>>(pxl_W,    wPxl,  (long)512 * 512);
  cvt_bf16_kernel<<<512, 256, 0, stream>>>(pxc_W,    wPxc,  (long)512 * 512);
  cvt_bf16_kernel<<<512, 256, 0, stream>>>(qry_W,    wQ,    (long)1536 * 512);
  cvt_bf16_kernel<<<512, 256, 0, stream>>>(key_W,    wK,    (long)1536 * 512);
  cvt_bf16_kernel<<<512, 256, 0, stream>>>(val_W,    wV,    (long)1536 * 512);
  cvt_bf16_kernel<<<512, 256, 0, stream>>>(mem_W,    wMem,  (long)1024 * 512);
  cvt_bf16_kernel<<<512, 256, 0, stream>>>(comb_W,   wComb, (long)1024 * 512);

  qbase_kernel<<<128, 256, 0, stream>>>(q_enc, qIn_W, qIn_b, qbase);
  norm_kernel<<<32768, 256, 0, stream>>>(images, Aimg);
  bcast_kernel<<<EW, 256, 0, stream>>>(initMem, xctx, M * 512);

  // x_loc = images_n @ initKB_W + b   -> f32 x_loc and bf16 into x_joint[:,0:512]
  gemm_launch(stream, Aimg, 1024, 0, wInit, 512, 0, 0, initKB_b,
              nullptr, 1, nullptr, 0, xloc, 512, xjnt, 1536, 0,
              (int)M, 512, 1024, 1, 1.0f);

  for (int t = 0; t < 4; ++t) {
    cmd_kernel<<<128, 256, 0, stream>>>(qbase, qIn2_W + (long)t * 512 * 512,
        qIn2_b + t * 512, lstm, qlen, c2l_W, c2l_b, pk_W, pk_b, pv_W, pv_b,
        sclK, sclV);

    pack_xctx_kernel<<<EW, 256, 0, stream>>>(xctx, xjnt, memin, M * 512);

    // proj_x_loc (f32, feeds the product epilogue)
    gemm_launch(stream, xjnt, 1536, 0, wPxl, 512, 0, 0, pxl_b,
                nullptr, 1, nullptr, 0, pxl, 512, nullptr, 0, 0,
                (int)M, 512, 512, 1, 1.0f);
    // proj_x_ctx * proj_x_loc -> bf16 into x_joint[:,1024:1536]
    gemm_launch(stream, xjnt + 512, 1536, 0, wPxc, 512, 0, 0, pxc_b,
                nullptr, 1, pxl, 512, nullptr, 0, xjnt + 1024, 1536, 0,
                (int)M, 512, 512, 1, 1.0f);

    // queries / keys / vals from x_joint (K = 1536); keys & vals gated by cmd proj
    gemm_launch(stream, xjnt, 1536, 0, wQ, 512, 0, 0, qry_b,
                nullptr, 1, nullptr, 0, nullptr, 0, Qb, 512, 0,
                (int)M, 512, 1536, 1, 1.0f);
    gemm_launch(stream, xjnt, 1536, 0, wK, 512, 0, 0, key_b,
                sclK, 256, nullptr, 0, nullptr, 0, Kb, 512, 0,
                (int)M, 512, 1536, 1, 1.0f);
    gemm_launch(stream, xjnt, 1536, 0, wV, 512, 0, 0, val_b,
                sclV, 256, nullptr, 0, nullptr, 0, Vb, 512, 0,
                (int)M, 512, 1536, 1, 1.0f);

    // edge scores: S[b] = Q[b] @ K[b]^T * 1/sqrt(512)   (batched, B^T async path)
    gemm_launch(stream, Qb, 512, 256L * 512, Kb, 512, 256L * 512, 1,
                nullptr, nullptr, 1, nullptr, 0, Sbuf, 256, nullptr, 0, 256L * 256,
                256, 256, 512, 128, SCALE);
    softmax_kernel<<<32768, 256, 0, stream>>>(Sbuf, Pbuf, ent);
    // message = P @ V  -> bf16 straight into mem_in[:,512:1024]
    gemm_launch(stream, Pbuf, 256, 256L * 256, Vb, 512, 256L * 512, 0,
                nullptr, nullptr, 1, nullptr, 0, nullptr, 0,
                memin + 512, 1024, 256L * 1024,
                256, 512, 256, 128, 1.0f);

    // x_ctx = concat(x_ctx, message) @ mem_update_W + b
    gemm_launch(stream, memin, 1024, 0, wMem, 512, 0, 0, mem_b,
                nullptr, 1, nullptr, 0, xctx, 512, nullptr, 0, 0,
                (int)M, 512, 1024, 1, 1.0f);
  }

  // out = concat(x_loc, x_ctx) @ combine_kb_W + b
  pack_final_kernel<<<EW, 256, 0, stream>>>(xloc, xctx, memin, M * 512);
  gemm_launch(stream, memin, 1024, 0, wComb, 512, 0, 0, comb_b,
              nullptr, 1, nullptr, 0, (float*)d_out, 512, nullptr, 0, 0,
              (int)M, 512, 1024, 1, 1.0f);
}
